// MultiheadAttention_78520592105818
// MI455X (gfx1250) — compile-verified, matrix operand order checked by reading, not run on hardware
//
#include <hip/hip_runtime.h>
#include <hip/hip_bf16.h>
#include <math.h>

typedef __bf16 bf16;
typedef __attribute__((ext_vector_type(16))) __bf16 v16bf;
typedef __attribute__((ext_vector_type(8)))  __bf16 v8bf;
typedef __attribute__((ext_vector_type(8)))  float  v8f;

#define B_   8
#define T_   4096
#define EMB_ 768
#define M_   384
#define EPS_ 1e-8f
#define KP   40   // LDS pitch (bf16) for [rows][32] K-chunk tiles; 80B rows keep 16B alignment

__device__ __forceinline__ v8f zero8() {
  v8f z = {0.f, 0.f, 0.f, 0.f, 0.f, 0.f, 0.f, 0.f};
  return z;
}

// 16x32 bf16 WMMA operand fragment from LDS tile [rows][KP] (row-major).
// ISA 7.12.2: lanes 0-15 -> rows 0-15 K=0..7,16..23; lanes 16-31 -> K=8..15,24..31.
__device__ __forceinline__ v16bf load_frag(const bf16* tile, int rowbase, int lane) {
  const int half = lane >> 4;
  const bf16* p = tile + (rowbase + (lane & 15)) * KP + half * 8;
  v8bf lo = *reinterpret_cast<const v8bf*>(p);
  v8bf hi = *reinterpret_cast<const v8bf*>(p + 16);
  v16bf f;
#pragma unroll
  for (int j = 0; j < 8; ++j) { f[j] = lo[j]; f[j + 8] = hi[j]; }
  return f;
}

__device__ __forceinline__ v8f wmma_bf16(v16bf a, v16bf b, v8f c) {
  return __builtin_amdgcn_wmma_f32_16x16x32_bf16(
      false, a, false, b, (short)0, c, false, false);
}

__device__ __forceinline__ void load8f(const float* p, float* r) {
  const float4* p4 = reinterpret_cast<const float4*>(p);
  float4 x0 = p4[0], x1 = p4[1];
  r[0] = x0.x; r[1] = x0.y; r[2] = x0.z; r[3] = x0.w;
  r[4] = x1.x; r[5] = x1.y; r[6] = x1.z; r[7] = x1.w;
}

// Async 16B global -> LDS copy (CDNA5, tracked by ASYNCcnt).
__device__ __forceinline__ void async_b128(unsigned lds_addr, const void* gaddr) {
  asm volatile("global_load_async_to_lds_b128 %0, %1, off"
               :: "v"(lds_addr), "v"(gaddr) : "memory");
}
__device__ __forceinline__ void wait_async0() {
  asm volatile("s_wait_asynccnt 0x0" ::: "memory");
}

// ---------------------------------------------------------------------------
// Kernel 1: xp[b,t,m] = exp(x[b,t]·w[m] - |x[b,t]|^2/2) / sqrt(M)   (bf16 out)
// Double-buffered LDS, register-staged (needs f32->bf16 cvt + row-norm).
// ---------------------------------------------------------------------------
__global__ void __launch_bounds__(256)
featurize_kernel(const float* __restrict__ x, const float* __restrict__ w,
                 bf16* __restrict__ xp)
{
  __shared__ __align__(16) bf16  At[2][64 * KP];
  __shared__ __align__(16) bf16  Bt[2][64 * KP];
  __shared__ float xds[64 * 4];
  __shared__ float xdv[64];

  const int tid = threadIdx.x, lane = tid & 31, wv = tid >> 5;
  const int mb = blockIdx.x * 64, tb = blockIdx.y * 64, b = blockIdx.z;
  const int srow = tid >> 2, scol = (tid & 3) * 8;
  const int tr = wv >> 1, tc = (wv & 1) * 2;
  const int so = srow * KP + scol;

  const float* arow = x + ((size_t)b * T_ + tb + srow) * EMB_ + scol;
  const float* brow = w + (size_t)(mb + srow) * EMB_ + scol;

  v8f c0 = zero8(), c1 = zero8();
  float xd_part = 0.f;

  {  // prologue: chunk 0 -> buf 0
    float av[8], bv[8];
    load8f(arow, av); load8f(brow, bv);
#pragma unroll
    for (int j = 0; j < 8; ++j) {
      xd_part += av[j] * av[j];
      At[0][so + j] = (bf16)av[j];
      Bt[0][so + j] = (bf16)bv[j];
    }
  }
  __syncthreads();

  const int NIT = EMB_ / 32;  // 24
  for (int it = 0; it < NIT; ++it) {
    const int cur = it & 1;
    const bool more = (it + 1) < NIT;
    float a2[8], b2[8];
    if (more) { load8f(arow + (it + 1) * 32, a2); load8f(brow + (it + 1) * 32, b2); }

    v16bf a  = load_frag(At[cur], tr * 16, lane);
    v16bf f0 = load_frag(Bt[cur], tc * 16, lane);
    v16bf f1 = load_frag(Bt[cur], (tc + 1) * 16, lane);
    c0 = wmma_bf16(a, f0, c0);
    c1 = wmma_bf16(a, f1, c1);

    if (more) {
      const int nxt = cur ^ 1;
#pragma unroll
      for (int j = 0; j < 8; ++j) {
        xd_part += a2[j] * a2[j];
        At[nxt][so + j] = (bf16)a2[j];
        Bt[nxt][so + j] = (bf16)b2[j];
      }
    }
    __syncthreads();
  }

  xds[srow * 4 + (tid & 3)] = xd_part;
  __syncthreads();
  if (tid < 64)
    xdv[tid] = 0.5f * (xds[tid * 4] + xds[tid * 4 + 1] +
                       xds[tid * 4 + 2] + xds[tid * 4 + 3]);
  __syncthreads();

  const float inv_sqrt_m = 0.05103103630798287f;  // 1/sqrt(384)
  const int half = lane >> 4, coll = lane & 15;
#pragma unroll
  for (int r = 0; r < 8; ++r) {
    int row = tr * 16 + r + 8 * half;
    float xd = xdv[row];
    size_t base = ((size_t)b * T_ + tb + row) * M_ + mb;
    xp[base + tc * 16 + coll]       = (bf16)(__expf(c0[r] - xd) * inv_sqrt_m);
    xp[base + (tc + 1) * 16 + coll] = (bf16)(__expf(c1[r] - xd) * inv_sqrt_m);
  }
}

// ---------------------------------------------------------------------------
// Kernel 2: ksum[b,m] = sum_t kp[b,t,m]
// ---------------------------------------------------------------------------
__global__ void __launch_bounds__(128)
colsum_kernel(const bf16* __restrict__ kp, float* __restrict__ ksum) {
  const int m = blockIdx.x * 128 + threadIdx.x;
  const int b = blockIdx.y;
  const bf16* p = kp + (size_t)b * T_ * M_ + m;
  float acc = 0.f;
  for (int t = 0; t < T_; ++t) acc += (float)p[(size_t)t * M_];
  ksum[b * M_ + m] = acc;
}

// ---------------------------------------------------------------------------
// Kernel 3: dinv[b,t] = 1 / (qp[b,t]·ksum[b] + eps)   (one wave32 per row)
// ---------------------------------------------------------------------------
__global__ void __launch_bounds__(256)
dinv_kernel(const bf16* __restrict__ qp, const float* __restrict__ ksum,
            float* __restrict__ dinv) {
  __shared__ float ks[M_];
  const int tid = threadIdx.x, lane = tid & 31, wv = tid >> 5;
  const int g0 = blockIdx.x * 8;
  const int b  = g0 / T_;
  for (int i = tid; i < M_; i += 256) ks[i] = ksum[b * M_ + i];
  __syncthreads();
  const int g = g0 + wv;
  const bf16* qr = qp + (size_t)g * M_;
  float acc = 0.f;
  for (int m = lane; m < M_; m += 32) acc += (float)qr[m] * ks[m];
#pragma unroll
  for (int s = 16; s > 0; s >>= 1) acc += __shfl_xor(acc, s, 32);
  if (lane == 0) dinv[g] = 1.f / (acc + EPS_);
}

// ---------------------------------------------------------------------------
// Kernel 4: kptv[b,n,m] = sum_t v[b,t,n] * kp[b,t,m]  (K=T, LDS transpose,
// double-buffered register staging)
// ---------------------------------------------------------------------------
__global__ void __launch_bounds__(256)
kptv_kernel(const float* __restrict__ v, const bf16* __restrict__ kp,
            bf16* __restrict__ kptv) {
  __shared__ __align__(16) bf16 Vt[2][64 * KP];  // rows = n, cols = t'
  __shared__ __align__(16) bf16 Kt[2][64 * KP];  // rows = m, cols = t'
  const int tid = threadIdx.x, lane = tid & 31, wv = tid >> 5;
  const int mb = blockIdx.x * 64, nb = blockIdx.y * 64, b = blockIdx.z;
  const int st = tid >> 3;        // t' 0..31
  const int sc = (tid & 7) * 8;   // col group
  const int tr = wv >> 1, tc = (wv & 1) * 2;

  const float* vrow = v + ((size_t)b * T_ + st) * EMB_ + nb + sc;
  const bf16*  krow = kp + ((size_t)b * T_ + st) * M_ + mb + sc;

  v8f c0 = zero8(), c1 = zero8();

  {  // prologue: t-chunk 0 -> buf 0
    float vv[8];
    load8f(vrow, vv);
    v8bf kv = *reinterpret_cast<const v8bf*>(krow);
#pragma unroll
    for (int j = 0; j < 8; ++j) {
      Vt[0][(sc + j) * KP + st] = (bf16)vv[j];
      Kt[0][(sc + j) * KP + st] = kv[j];
    }
  }
  __syncthreads();

  const int NIT = T_ / 32;  // 128
  for (int it = 0; it < NIT; ++it) {
    const int cur = it & 1;
    const bool more = (it + 1) < NIT;
    float v2[8]; v8bf k2;
    if (more) {
      load8f(vrow + (size_t)(it + 1) * 32 * EMB_, v2);
      k2 = *reinterpret_cast<const v8bf*>(krow + (size_t)(it + 1) * 32 * M_);
    }

    v16bf a  = load_frag(Vt[cur], tr * 16, lane);
    v16bf f0 = load_frag(Kt[cur], tc * 16, lane);
    v16bf f1 = load_frag(Kt[cur], (tc + 1) * 16, lane);
    c0 = wmma_bf16(a, f0, c0);
    c1 = wmma_bf16(a, f1, c1);

    if (more) {
      const int nxt = cur ^ 1;
#pragma unroll
      for (int j = 0; j < 8; ++j) {
        Vt[nxt][(sc + j) * KP + st] = (bf16)v2[j];
        Kt[nxt][(sc + j) * KP + st] = k2[j];
      }
    }
    __syncthreads();
  }

  const int half = lane >> 4, coll = lane & 15;
#pragma unroll
  for (int r = 0; r < 8; ++r) {
    int n = tr * 16 + r + 8 * half;
    size_t base = ((size_t)b * EMB_ + nb + n) * M_ + mb;
    kptv[base + tc * 16 + coll]       = (bf16)c0[r];
    kptv[base + (tc + 1) * 16 + coll] = (bf16)c1[r];
  }
}

// ---------------------------------------------------------------------------
// Kernel 5: y[b,t,n] = (qp[b,t]·kptv[b,n]) * dinv[b,t]   (bf16 out)
// Both operands bf16 in memory -> async global->LDS copies (ASYNCcnt),
// double-buffered, one barrier per K-chunk.
// ---------------------------------------------------------------------------
__global__ void __launch_bounds__(256)
ycompute_kernel(const bf16* __restrict__ qp, const bf16* __restrict__ kptv,
                const float* __restrict__ dinv, bf16* __restrict__ y) {
  __shared__ __align__(16) bf16  At[2][64 * KP];
  __shared__ __align__(16) bf16  Bt[2][64 * KP];
  __shared__ float dv[64];
  const int tid = threadIdx.x, lane = tid & 31, wv = tid >> 5;
  const int nb = blockIdx.x * 64, tb = blockIdx.y * 64, b = blockIdx.z;
  const int srow = tid >> 2, scol = (tid & 3) * 8;
  const int tr = wv >> 1, tc = (wv & 1) * 2;

  const bf16* arow = qp + ((size_t)b * T_ + tb + srow) * M_ + scol;
  const bf16* brow = kptv + ((size_t)b * EMB_ + nb + srow) * M_ + scol;

  const unsigned eo = (unsigned)((srow * KP + scol) * 2);
  const unsigned aL[2] = {(unsigned)(uintptr_t)&At[0][0] + eo,
                          (unsigned)(uintptr_t)&At[1][0] + eo};
  const unsigned bL[2] = {(unsigned)(uintptr_t)&Bt[0][0] + eo,
                          (unsigned)(uintptr_t)&Bt[1][0] + eo};

  if (tid < 64) dv[tid] = dinv[b * T_ + tb + tid];

  async_b128(aL[0], arow);
  async_b128(bL[0], brow);
  v8f c0 = zero8(), c1 = zero8();
  wait_async0();
  __syncthreads();

  const int NIT = M_ / 32;  // 12
  for (int it = 0; it < NIT; ++it) {
    const int cur = it & 1;
    if (it + 1 < NIT) {                      // prefetch next chunk into other buffer
      async_b128(aL[cur ^ 1], arow + (it + 1) * 32);
      async_b128(bL[cur ^ 1], brow + (it + 1) * 32);
    }
    v16bf a  = load_frag(At[cur], tr * 16, lane);
    v16bf f0 = load_frag(Bt[cur], tc * 16, lane);
    v16bf f1 = load_frag(Bt[cur], (tc + 1) * 16, lane);
    c0 = wmma_bf16(a, f0, c0);
    c1 = wmma_bf16(a, f1, c1);
    wait_async0();
    __syncthreads();
  }

  const int half = lane >> 4, coll = lane & 15;
#pragma unroll
  for (int r = 0; r < 8; ++r) {
    int row = tr * 16 + r + 8 * half;
    float s = dv[row];
    size_t base = ((size_t)b * T_ + tb + row) * EMB_ + nb;
    y[base + tc * 16 + coll]       = (bf16)(c0[r] * s);
    y[base + (tc + 1) * 16 + coll] = (bf16)(c1[r] * s);
  }
}

// ---------------------------------------------------------------------------
// Kernel 6: out[b,t,e] = sum_n y[b,t,n] * proj_W[e,n]   (f32 out)
// y side: async global->LDS (bf16); proj_W side: register-staged cvt.
// ---------------------------------------------------------------------------
__global__ void __launch_bounds__(256)
proj_kernel(const bf16* __restrict__ y, const float* __restrict__ pw,
            float* __restrict__ out) {
  __shared__ __align__(16) bf16 At[2][64 * KP];
  __shared__ __align__(16) bf16 Bt[2][64 * KP];
  const int tid = threadIdx.x, lane = tid & 31, wv = tid >> 5;
  const int eb = blockIdx.x * 64, tb = blockIdx.y * 64, b = blockIdx.z;
  const int srow = tid >> 2, scol = (tid & 3) * 8;
  const int tr = wv >> 1, tc = (wv & 1) * 2;
  const int so = srow * KP + scol;

  const bf16*  arow = y + ((size_t)b * T_ + tb + srow) * EMB_ + scol;
  const float* brow = pw + (size_t)(eb + srow) * EMB_ + scol;

  const unsigned eo = (unsigned)(so * 2);
  const unsigned aL[2] = {(unsigned)(uintptr_t)&At[0][0] + eo,
                          (unsigned)(uintptr_t)&At[1][0] + eo};

  async_b128(aL[0], arow);
  {
    float bv[8];
    load8f(brow, bv);
#pragma unroll
    for (int j = 0; j < 8; ++j) Bt[0][so + j] = (bf16)bv[j];
  }
  v8f c0 = zero8(), c1 = zero8();
  wait_async0();
  __syncthreads();

  const int NIT = EMB_ / 32;  // 24
  for (int it = 0; it < NIT; ++it) {
    const int cur = it & 1;
    const bool more = (it + 1) < NIT;
    float b2[8];
    if (more) {
      async_b128(aL[cur ^ 1], arow + (it + 1) * 32);
      load8f(brow + (it + 1) * 32, b2);
    }
    v16bf a  = load_frag(At[cur], tr * 16, lane);
    v16bf f0 = load_frag(Bt[cur], tc * 16, lane);
    v16bf f1 = load_frag(Bt[cur], (tc + 1) * 16, lane);
    c0 = wmma_bf16(a, f0, c0);
    c1 = wmma_bf16(a, f1, c1);
    if (more) {
      const int nxt = cur ^ 1;
#pragma unroll
      for (int j = 0; j < 8; ++j) Bt[nxt][so + j] = (bf16)b2[j];
    }
    wait_async0();
    __syncthreads();
  }

  const int half = lane >> 4, coll = lane & 15;
#pragma unroll
  for (int r = 0; r < 8; ++r) {
    int row = tr * 16 + r + 8 * half;
    size_t base = ((size_t)b * T_ + tb + row) * EMB_ + eb;
    out[base + tc * 16 + coll]       = c0[r];
    out[base + (tc + 1) * 16 + coll] = c1[r];
  }
}

// ---------------------------------------------------------------------------
extern "C" void kernel_launch(void* const* d_in, const int* in_sizes, int n_in,
                              void* d_out, int out_size, void* d_ws, size_t ws_size,
                              hipStream_t stream) {
  (void)in_sizes; (void)n_in; (void)out_size; (void)ws_size;
  const float* q  = (const float*)d_in[0];
  const float* k  = (const float*)d_in[1];
  const float* v  = (const float*)d_in[2];
  const float* w  = (const float*)d_in[3];
  const float* pw = (const float*)d_in[4];
  float* out = (float*)d_out;

  char* ws = (char*)d_ws;
  bf16*  qp    = (bf16*)(ws + 0);            // B*T*M bf16   = 25,165,824
  bf16*  kp    = (bf16*)(ws + 25165824);     // B*T*M bf16   = 25,165,824
  float* ksum  = (float*)(ws + 50331648);    // B*M f32      =     12,288
  bf16*  kptv  = (bf16*)(ws + 50343936);     // B*EMB*M bf16 =  4,718,592
  float* dinv  = (float*)(ws + 55062528);    // B*T f32      =    131,072
  bf16*  y     = (bf16*)(ws + 55193600);     // B*T*EMB bf16 = 50,331,648

  dim3 blk(256);
  featurize_kernel<<<dim3(M_ / 64, T_ / 64, B_), blk, 0, stream>>>(q, w, qp);
  featurize_kernel<<<dim3(M_ / 64, T_ / 64, B_), blk, 0, stream>>>(k, w, kp);
  colsum_kernel<<<dim3(M_ / 128, B_), dim3(128), 0, stream>>>(kp, ksum);
  dinv_kernel<<<dim3((B_ * T_) / 8), blk, 0, stream>>>(qp, ksum, dinv);
  kptv_kernel<<<dim3(M_ / 64, EMB_ / 64, B_), blk, 0, stream>>>(v, kp, kptv);
  ycompute_kernel<<<dim3(EMB_ / 64, T_ / 64, B_), blk, 0, stream>>>(qp, kptv, dinv, y);
  proj_kernel<<<dim3(EMB_ / 64, T_ / 64, B_), blk, 0, stream>>>(y, pw, out);
}
